// AVDFullLinearMix_25409026523191
// MI455X (gfx1250) — matrix operand order repaired and checked by reading, not compiled, not run on hardware
//
#include <hip/hip_runtime.h>

#define DIM 256
#define TOK 16          // tokens per workgroup (WMMA M/N tile = 16)
#define PADC 264        // 256 + 8 bf16 pad -> row stride 528B, 4-bank skew per row
#define THREADS 256     // 8 wave32 waves
#define NPW 32          // output channels owned per wave (2 n-tiles)

typedef __attribute__((ext_vector_type(16))) __bf16 bf16x16;
typedef __attribute__((ext_vector_type(8)))  float  v8f;

// Hardware packed f32->bf16 (RNE): one VALU op per two elements.
__device__ __forceinline__ unsigned int cvt_pk_bf16(float a, float b) {
  unsigned int r;
  asm("v_cvt_pk_bf16_f32 %0, %1, %2" : "=v"(r) : "v"(a), "v"(b));
  return r;
}
__device__ __forceinline__ unsigned short f2bf(float f) {
  return (unsigned short)cvt_pk_bf16(f, f);
}

// A fragment (16x32 bf16): lane = row M (lane&15); lane<16 holds K={k0..k0+7, k0+16..k0+23},
// lane>=16 holds K={k0+8..k0+15, k0+24..k0+31}  (ISA 7.12.2 16-bit A table)
__device__ __forceinline__ bf16x16 load_afrag(const unsigned short* row, int k0, int hi) {
  union { bf16x16 v; uint4 q[2]; } u;
  u.q[0] = *(const uint4*)(row + k0 + hi * 8);
  u.q[1] = *(const uint4*)(row + k0 + 16 + hi * 8);
  return u.v;
}

// B fragment (32x16 bf16): lane = col N (output channel); lane<16 holds K={k0..k0+7, k0+16..k0+23},
// lane>=16 holds K={k0+8..k0+15, k0+24..k0+31} (matches A fragment K order).
// B[k, o] = W[o, k]  ->  two 8-float runs of W's row, packed-converted to bf16.
__device__ __forceinline__ bf16x16 load_bfrag(const float* Wrow, int k0, int hi) {
  const float4* p = (const float4*)(Wrow + k0 + hi * 8);
  union { bf16x16 v; unsigned int d[8]; } u;
  float4 f0 = p[0], f1 = p[1], f2 = p[4], f3 = p[5];
  u.d[0] = cvt_pk_bf16(f0.x, f0.y); u.d[1] = cvt_pk_bf16(f0.z, f0.w);
  u.d[2] = cvt_pk_bf16(f1.x, f1.y); u.d[3] = cvt_pk_bf16(f1.z, f1.w);
  u.d[4] = cvt_pk_bf16(f2.x, f2.y); u.d[5] = cvt_pk_bf16(f2.z, f2.w);
  u.d[6] = cvt_pk_bf16(f3.x, f3.y); u.d[7] = cvt_pk_bf16(f3.z, f3.w);
  return u.v;
}

// Convert a wave's weight-row slice once into 8 cached fragments (64 VGPRs).
__device__ __forceinline__ void load_bfrags(bf16x16* bw, const float* Wrow, int hi) {
#pragma unroll
  for (int ks = 0; ks < 8; ++ks) bw[ks] = load_bfrag(Wrow, ks * 32, hi);
}

// Full K=256 GEMM pass with register-cached B fragments.
__device__ __forceinline__ v8f gemmK256(const unsigned short* Arow, const bf16x16* bw,
                                        int hi, v8f acc) {
#pragma unroll
  for (int ks = 0; ks < 8; ++ks) {
    bf16x16 a = load_afrag(Arow, ks * 32, hi);
    acc = __builtin_amdgcn_wmma_f32_16x16x32_bf16(false, a, false, bw[ks], (short)0, acc,
                                                  false, false);
  }
  return acc;
}

__global__ __launch_bounds__(THREADS, 2)
void avd_mix_kernel(const float* __restrict__ xa, const float* __restrict__ xv,
                    const float* __restrict__ xd,
                    const float* __restrict__ Wtt, const float* __restrict__ Wtr,
                    const float* __restrict__ Wtd, const float* __restrict__ Wvd,
                    const float* __restrict__ Wdv,
                    float* __restrict__ out_a, float* __restrict__ out_v,
                    float* __restrict__ out_d, int n_tiles) {
  __shared__ unsigned short xd_s[9][TOK][PADC];   // [spatial i*3+j][token][channel]
  __shared__ unsigned short xv_s[3][TOK][PADC];   // x_v inputs (kept intact)
  __shared__ unsigned short xa_s[TOK][PADC];      // x_a inputs (kept intact)
  __shared__ unsigned short tr_s[TOK][PADC];      // trace(x_d1) for TensTrace GEMM
  __shared__ unsigned short w_s[3][TOK][PADC];    // curl(x_d1) for ChiralMix d->v GEMM

  const int tid = threadIdx.x;
  const int T0  = blockIdx.x * TOK;

  // ---------------- Phase 0: stage tile into LDS as bf16 (coalesced f32 reads) ----
  {
    const float* g = xd + (size_t)T0 * (DIM * 9);
    for (int idx = tid; idx < TOK * DIM * 9; idx += THREADS) {
      int t = idx / (DIM * 9);
      int r = idx - t * (DIM * 9);
      int c = r / 9, s = r - c * 9;
      xd_s[s][t][c] = f2bf(g[idx]);
    }
    const float* gv = xv + (size_t)T0 * (DIM * 3);
    for (int idx = tid; idx < TOK * DIM * 3; idx += THREADS) {
      int t = idx / (DIM * 3);
      int r = idx - t * (DIM * 3);
      int c = r / 3, i = r - c * 3;
      xv_s[i][t][c] = f2bf(gv[idx]);
    }
    const float* ga = xa + (size_t)T0 * DIM;
    for (int idx = tid; idx < TOK * DIM; idx += THREADS) {
      int t = idx / DIM, c = idx - t * DIM;
      xa_s[t][c] = f2bf(ga[idx]);
    }
    if (blockIdx.x + 1 < (unsigned)n_tiles)
      __builtin_prefetch(xd + (size_t)(T0 + TOK) * (DIM * 9), 0, 1);
  }
  __syncthreads();

  const int lane = tid & 31;
  const int wv   = tid >> 5;
  const int tl   = lane & 15;   // A row (token) / B col (channel) within tile
  const int hi   = lane >> 4;   // lane half selects K sub-ranges

  bf16x16 bw[8];                // register-cached B fragments for current weight slice

  // ---------------- Phase 1 (fused per n-tile): delta, g, TensTrans + epilogues ---
  // x_d_out[t,o,i,j] = xd + t1 + delta*d_ij + vd(i,j),  vd(i,j) = g[k]*e[k,j,i]
  // tr[t,o]   = sum_i (xd + t1)[i,i]    (delta excluded: trace precedes TensDelta)
  // w[t,o,k]  = e[i,j,k]*(xd + t1)[i,j] (delta/vd drop: diag-only / added after)
#pragma unroll 1
  for (int nt = 0; nt < 2; ++nt) {
    const int o = wv * NPW + nt * 16 + tl;
    const v8f z = 0.0f;

    load_bfrags(bw, Wtd + (size_t)o * DIM, hi);
    v8f delta = gemmK256(&xa_s[tl][0], bw, hi, z);

    v8f g[3];
    load_bfrags(bw, Wvd + (size_t)o * DIM, hi);   // converted once, used 3x
#pragma unroll
    for (int i = 0; i < 3; ++i)
      g[i] = gemmK256(&xv_s[i][tl][0], bw, hi, z);

    load_bfrags(bw, Wtt + (size_t)o * DIM, hi);   // converted once, used 9x
    v8f tr = 0.0f;
    v8f wc[3];
#pragma unroll
    for (int k = 0; k < 3; ++k) wc[k] = 0.0f;

#pragma unroll
    for (int i = 0; i < 3; ++i) {
#pragma unroll
      for (int j = 0; j < 3; ++j) {
        // t1[:,:,i,j] needs A[t,c] = x_d[t,c,j,i]  -> plane j*3+i
        v8f acc = gemmK256(&xd_s[j * 3 + i][tl][0], bw, hi, z);
        if (i == j) {
#pragma unroll
          for (int r = 0; r < 8; ++r) {
            size_t idx = (size_t)(T0 + r + hi * 8) * (DIM * 9) + o * 9 + i * 3 + j;
            float base = xd[idx] + acc[r];     // f32 residual re-read (L2-hot)
            tr[r] += base;
            out_d[idx] = base + delta[r];
          }
        } else {
          const int   kk  = 3 - i - j;
          const float sgn = ((j - i + 3) % 3 == 1) ? 1.0f : -1.0f;  // e[i,j,kk]
#pragma unroll
          for (int r = 0; r < 8; ++r) {
            size_t idx = (size_t)(T0 + r + hi * 8) * (DIM * 9) + o * 9 + i * 3 + j;
            float base = xd[idx] + acc[r];
            wc[kk][r] += sgn * base;
            out_d[idx] = base - sgn * g[kk][r];  // e[kk,j,i] = -e[i,j,kk]
          }
        }
      }
    }
    // scatter tr / curl into LDS planes (D-layout: lane = channel, vgpr = token)
#pragma unroll
    for (int r = 0; r < 8; ++r) {
      tr_s[r + hi * 8][o] = f2bf(tr[r]);
#pragma unroll
      for (int k = 0; k < 3; ++k)
        w_s[k][r + hi * 8][o] = f2bf(wc[k][r]);
    }
    __builtin_amdgcn_sched_barrier(0);  // keep n-tile iterations from interleaving
  }
  __syncthreads();

  // ---------------- Phase 2: TensTrace -> x_a_out ; ChiralMix d->v -> x_v_out -----
#pragma unroll 1
  for (int nt = 0; nt < 2; ++nt) {
    const int o = wv * NPW + nt * 16 + tl;
    const v8f z = 0.0f;

    load_bfrags(bw, Wtr + (size_t)o * DIM, hi);
    v8f accA = gemmK256(&tr_s[tl][0], bw, hi, z);
#pragma unroll
    for (int r = 0; r < 8; ++r) {
      size_t idx = (size_t)(T0 + r + hi * 8) * DIM + o;
      out_a[idx] = xa[idx] + accA[r];
    }

    load_bfrags(bw, Wdv + (size_t)o * DIM, hi);   // converted once, used 3x
#pragma unroll
    for (int k = 0; k < 3; ++k) {
      v8f accV = gemmK256(&w_s[k][tl][0], bw, hi, z);
#pragma unroll
      for (int r = 0; r < 8; ++r) {
        size_t idx = (size_t)(T0 + r + hi * 8) * (DIM * 3) + o * 3 + k;
        out_v[idx] = xv[idx] + accV[r];
      }
    }
    __builtin_amdgcn_sched_barrier(0);
  }
}

extern "C" void kernel_launch(void* const* d_in, const int* in_sizes, int n_in,
                              void* d_out, int out_size, void* d_ws, size_t ws_size,
                              hipStream_t stream) {
  const float* xa  = (const float*)d_in[0];
  const float* xv  = (const float*)d_in[1];
  const float* xd  = (const float*)d_in[2];
  const float* Wtt = (const float*)d_in[3];
  const float* Wtr = (const float*)d_in[4];
  const float* Wtd = (const float*)d_in[5];
  const float* Wvd = (const float*)d_in[6];
  const float* Wdv = (const float*)d_in[7];

  const int BN = in_sizes[0] / DIM;             // 16384 tokens
  float* out   = (float*)d_out;
  float* out_a = out;
  float* out_v = out + (size_t)BN * DIM;        // x_v block after x_a block
  float* out_d = out_v + (size_t)BN * DIM * 3;  // x_d block last

  const int n_tiles = BN / TOK;                 // 1024 workgroups
  avd_mix_kernel<<<dim3(n_tiles), dim3(THREADS), 0, stream>>>(
      xa, xv, xd, Wtt, Wtr, Wtd, Wvd, Wdv, out_a, out_v, out_d, n_tiles);
}
